// Seq2Seq_54528904790152
// MI455X (gfx1250) — matrix operand order repaired and checked by reading, not compiled, not run on hardware
//
#include <hip/hip_runtime.h>
#include <hip/hip_bf16.h>

typedef __attribute__((ext_vector_type(16))) _Float16 v16h;
typedef __attribute__((ext_vector_type(8)))  float    v8f;

// Problem constants (match reference)
constexpr int NB  = 1024;   // batch
constexpr int NL  = 512;    // encoder sequence length
constexpr int NT  = 96;     // decoder steps
constexpr int HE  = 128;    // encoder hidden
constexpr int HD  = 256;    // decoder hidden
constexpr int MT  = 16;     // batch tile per workgroup

// ---------------------------------------------------------------------------
// Activations: use CDNA5 hardware V_TANH_F32 when the builtin exists.
#if defined(__has_builtin)
#  if __has_builtin(__builtin_amdgcn_tanhf)
#    define HW_TANH 1
__device__ __forceinline__ float htanh(float x) { return __builtin_amdgcn_tanhf(x); }
#  elif __has_builtin(__builtin_amdgcn_tanh_f32)
#    define HW_TANH 1
__device__ __forceinline__ float htanh(float x) { return __builtin_amdgcn_tanh_f32(x); }
#  endif
#endif
#ifndef HW_TANH
// Branch-free fallback: tanh(x) = 1 - 2/(exp(2x)+1) with fast exp + raw rcp.
__device__ __forceinline__ float htanh(float x) {
    float e = __expf(2.0f * x);
    return 1.0f - 2.0f * __builtin_amdgcn_rcpf(e + 1.0f);
}
#endif
__device__ __forceinline__ float sigf(float x) { return 0.5f + 0.5f * htanh(0.5f * x); }

// ---------------------------------------------------------------------------
// h-fragment LDS layout helper: element (m, k) of the f16 A-matrix tile is
// stored so a wave can load v16h per lane per 32-wide K chunk:
//   lane = m | ((k>>4)&1)<<4 ;  half j = k&15 ; chunk = k>>5
__device__ __forceinline__ int hfrag_idx(int m, int k) {
    int kc = k >> 5, r = k & 31;
    int lane = m + ((r >> 4) << 4);
    int jj = r & 15;
    return (kc * 32 + lane) * 16 + jj;
}

// ---------------------------------------------------------------------------
// Prep: convert decoder Whh [1024,256] f32 -> f16, B-fragment-major in ws.
// Fragment (nt 0..63, kc 0..7): lane l holds column n = nt*16 + (l&15),
// rows k = kc*32 + (l>>4)*16 + j  (j = 0..15), 32B contiguous per lane.
__global__ __launch_bounds__(256) void seq2seq_prep_w(const float* __restrict__ Whh,
                                                      _Float16* __restrict__ wsw) {
    int gid = blockIdx.x * 256 + threadIdx.x;     // 0 .. 262143
    int j  = gid & 15;
    int l  = (gid >> 4) & 31;
    int kc = (gid >> 9) & 7;
    int nt = gid >> 12;
    int n = nt * 16 + (l & 15);
    int k = kc * 32 + (l >> 4) * 16 + j;
    wsw[gid] = (_Float16)Whh[n * HD + k];
}

// ---------------------------------------------------------------------------
// Encoder: grid (64 batch tiles, 2 directions), 256 threads (8 waves).
// Wave w owns gate columns n = g*128 + w*16 + (lane&15) for g in {i,f,g,o}.
__global__ __launch_bounds__(256) void seq2seq_encoder(
    const float* __restrict__ src,
    const float* __restrict__ Wih_f, const float* __restrict__ Whh_f, const float* __restrict__ b_f,
    const float* __restrict__ Wih_b, const float* __restrict__ Whh_b, const float* __restrict__ b_b,
    float* __restrict__ ws) {
    const int bt  = blockIdx.x;
    const int dir = blockIdx.y;
    const int tid = threadIdx.x;
    const int w   = tid >> 5;
    const int l   = tid & 31;
    const int nrel = l & 15;
    const int khal = l >> 4;

    const float* Wih = dir ? Wih_b : Wih_f;
    const float* Whh = dir ? Whh_b : Whh_f;
    const float* bi  = dir ? b_b   : b_f;
    float* h_ws = ws + (dir ? 2 * NB * HE : 0);
    float* c_ws = h_ws + NB * HE;

    __shared__ float x_lds[MT * NL];                       // 32 KB
    __shared__ alignas(32) _Float16 h_frag[4 * 32 * 16];   // 4 KB, A-frag layout

    for (int i = tid; i < MT * NL; i += 256) {
        int m = i / NL, t = i % NL;
        x_lds[i] = src[(bt * MT + m) * NL + t];
    }
    for (int i = tid; i < 4 * 32 * 16; i += 256) h_frag[i] = (_Float16)0.0f;

    // Weight B-fragments in registers: 4 gates x 4 K-chunks.
    v16h wfr[4][4];
#pragma unroll
    for (int g = 0; g < 4; ++g) {
        int n = g * HE + w * 16 + nrel;
#pragma unroll
        for (int kc = 0; kc < 4; ++kc) {
            const float* row = Whh + n * HE + kc * 32 + khal * 16;
            v16h t;
#pragma unroll
            for (int j = 0; j < 16; ++j) t[j] = (_Float16)row[j];
            wfr[g][kc] = t;
        }
    }
    float wxv[4], bv[4];
#pragma unroll
    for (int g = 0; g < 4; ++g) {
        int n = g * HE + w * 16 + nrel;
        wxv[g] = Wih[n];
        bv[g]  = bi[n];
    }

    float creg[8];
#pragma unroll
    for (int j = 0; j < 8; ++j) creg[j] = 0.0f;

    const int hid = w * 16 + nrel;   // this lane's hidden index
    __syncthreads();

    for (int t = 0; t < NL; ++t) {
        const int xt = dir ? (NL - 1 - t) : t;

        v16h afr[4];
#pragma unroll
        for (int kc = 0; kc < 4; ++kc)
            afr[kc] = *(const v16h*)(&h_frag[(kc * 32 + l) * 16]);
        float xv[8];
#pragma unroll
        for (int j = 0; j < 8; ++j) xv[j] = x_lds[(j + 8 * khal) * NL + xt];
        __syncthreads();   // all waves done reading previous h

        v8f acc[4];
#pragma unroll
        for (int g = 0; g < 4; ++g) {
            v8f a = {};
#pragma unroll
            for (int kc = 0; kc < 4; ++kc)
                a = __builtin_amdgcn_wmma_f32_16x16x32_f16(false, afr[kc], false,
                                                           wfr[g][kc], (short)0, a,
                                                           false, false);
            acc[g] = a;
        }

#pragma unroll
        for (int j = 0; j < 8; ++j) {
            int m = j + 8 * khal;
            float gi = acc[0][j] + xv[j] * wxv[0] + bv[0];
            float gf = acc[1][j] + xv[j] * wxv[1] + bv[1];
            float gg = acc[2][j] + xv[j] * wxv[2] + bv[2];
            float go = acc[3][j] + xv[j] * wxv[3] + bv[3];
            float c = sigf(gf) * creg[j] + sigf(gi) * htanh(gg);
            float h = sigf(go) * htanh(c);
            creg[j] = c;
            h_frag[hfrag_idx(m, hid)] = (_Float16)h;
            if (t == NL - 1) {
                h_ws[(bt * MT + m) * HE + hid] = h;
                c_ws[(bt * MT + m) * HE + hid] = c;
            }
        }
        __syncthreads();   // new h visible
    }
}

// ---------------------------------------------------------------------------
// Decoder: grid 64 batch tiles, 256 threads (8 waves).
// Wave w owns hidden slice [w*32, w*32+32): n-tiles g*16 + w*2 + {0,1}.
__global__ __launch_bounds__(256) void seq2seq_decoder(
    const float* __restrict__ dWih, const float* __restrict__ db,
    const float* __restrict__ oW, const float* __restrict__ ob,
    const float* __restrict__ ws, const _Float16* __restrict__ wsw,
    float* __restrict__ out) {
    const int bt  = blockIdx.x;
    const int tid = threadIdx.x;
    const int w   = tid >> 5;
    const int l   = tid & 31;
    const int nrel = l & 15;
    const int khal = l >> 4;

    __shared__ alignas(32) _Float16 h_frag[8 * 32 * 16];   // 8 KB
    __shared__ float h32[MT * HD];                          // 16 KB
    __shared__ float red[16 * 16];
    __shared__ float y_lds[MT];
    __shared__ float ow[HD];

    const float* h_f = ws;
    const float* c_f = ws + NB * HE;
    const float* h_b = ws + 2 * NB * HE;
    const float* c_b = ws + 3 * NB * HE;

    for (int e = tid; e < MT * HD; e += 256) {
        int m = e >> 8, hid = e & 255;
        float v = (hid < HE) ? h_f[(bt * MT + m) * HE + hid]
                             : h_b[(bt * MT + m) * HE + hid - HE];
        h_frag[hfrag_idx(m, hid)] = (_Float16)v;
    }
    for (int i = tid; i < HD; i += 256) ow[i] = oW[i];
    if (tid < MT) y_lds[tid] = 0.0f;

    float creg[2][8];
#pragma unroll
    for (int p = 0; p < 2; ++p) {
        int hid = w * 32 + p * 16 + nrel;
#pragma unroll
        for (int j = 0; j < 8; ++j) {
            int m = j + 8 * khal;
            creg[p][j] = (hid < HE) ? c_f[(bt * MT + m) * HE + hid]
                                    : c_b[(bt * MT + m) * HE + hid - HE];
        }
    }
    float wxv[4][2], bv[4][2];
#pragma unroll
    for (int g = 0; g < 4; ++g)
#pragma unroll
        for (int p = 0; p < 2; ++p) {
            int n = g * HD + w * 32 + p * 16 + nrel;
            wxv[g][p] = dWih[n];
            bv[g][p]  = db[n];
        }
    const float outb = ob[0];
    __syncthreads();

    for (int t = 0; t < NT; ++t) {
        v16h afr[8];
#pragma unroll
        for (int kc = 0; kc < 8; ++kc)
            afr[kc] = *(const v16h*)(&h_frag[(kc * 32 + l) * 16]);
        float yv[8];
#pragma unroll
        for (int j = 0; j < 8; ++j) yv[j] = y_lds[j + 8 * khal];
        __syncthreads();

        v8f acc[4][2];
#pragma unroll
        for (int g = 0; g < 4; ++g)
#pragma unroll
            for (int p = 0; p < 2; ++p) {
                int nt = g * 16 + w * 2 + p;
                v8f a = {};
#pragma unroll
                for (int kc = 0; kc < 8; ++kc) {
                    v16h bf = *(const v16h*)(&wsw[((nt * 8 + kc) * 32 + l) * 16]);
                    a = __builtin_amdgcn_wmma_f32_16x16x32_f16(false, afr[kc], false,
                                                               bf, (short)0, a,
                                                               false, false);
                }
                acc[g][p] = a;
            }

#pragma unroll
        for (int p = 0; p < 2; ++p) {
            int hid = w * 32 + p * 16 + nrel;
#pragma unroll
            for (int j = 0; j < 8; ++j) {
                int m = j + 8 * khal;
                float gi = acc[0][p][j] + yv[j] * wxv[0][p] + bv[0][p];
                float gf = acc[1][p][j] + yv[j] * wxv[1][p] + bv[1][p];
                float gg = acc[2][p][j] + yv[j] * wxv[2][p] + bv[2][p];
                float go = acc[3][p][j] + yv[j] * wxv[3][p] + bv[3][p];
                float c = sigf(gf) * creg[p][j] + sigf(gi) * htanh(gg);
                float h = sigf(go) * htanh(c);
                creg[p][j] = c;
                h_frag[hfrag_idx(m, hid)] = (_Float16)h;
                h32[m * HD + hid] = h;
            }
        }
        __syncthreads();

        // out projection: y[m] = sum_k h[m,k]*oW[k] + ob
        {
            int m = tid >> 4, part = tid & 15;
            float s = 0.0f;
#pragma unroll
            for (int k = 0; k < 16; ++k)
                s += h32[m * HD + part * 16 + k] * ow[part * 16 + k];
            red[tid] = s;
        }
        __syncthreads();
        if (tid < MT) {
            float s = outb;
#pragma unroll
            for (int k = 0; k < 16; ++k) s += red[tid * 16 + k];
            y_lds[tid] = s;
            out[(bt * MT + tid) * NT + t] = s;
        }
        __syncthreads();
    }
}

// ---------------------------------------------------------------------------
extern "C" void kernel_launch(void* const* d_in, const int* in_sizes, int n_in,
                              void* d_out, int out_size, void* d_ws, size_t ws_size,
                              hipStream_t stream) {
    (void)in_sizes; (void)n_in; (void)out_size; (void)ws_size;
    const float* src    = (const float*)d_in[0];
    const float* eWih_f = (const float*)d_in[2];
    const float* eWhh_f = (const float*)d_in[3];
    const float* eb_f   = (const float*)d_in[4];
    const float* eWih_b = (const float*)d_in[5];
    const float* eWhh_b = (const float*)d_in[6];
    const float* eb_b   = (const float*)d_in[7];
    const float* dWih   = (const float*)d_in[8];
    const float* dWhh   = (const float*)d_in[9];
    const float* db     = (const float*)d_in[10];
    const float* oW     = (const float*)d_in[11];
    const float* ob     = (const float*)d_in[12];
    float* out = (float*)d_out;

    float* wsf = (float*)d_ws;                         // 4 * 1024*128 f32 enc states
    _Float16* wsw = (_Float16*)(wsf + 4 * NB * HE);    // 1024*256 f16 dec weights

    seq2seq_prep_w<<<dim3((64 * 8 * 32 * 16) / 256), 256, 0, stream>>>(dWhh, wsw);
    seq2seq_encoder<<<dim3(NB / MT, 2), 256, 0, stream>>>(
        src, eWih_f, eWhh_f, eb_f, eWih_b, eWhh_b, eb_b, wsf);
    seq2seq_decoder<<<dim3(NB / MT), 256, 0, stream>>>(dWih, db, oW, ob, wsf, wsw, out);
}
